// Attention_69965017252153
// MI455X (gfx1250) — matrix operand order repaired and checked by reading, not compiled, not run on hardware
//
#include <hip/hip_runtime.h>
#include <hip/hip_bf16.h>

// ---------------------------------------------------------------------------
// Multi-head causal attention, B=4 S=2048 D=768 H=4 DH=192, fp32 in/out.
// bf16 WMMA (v_wmma_f32_16x16x32_bf16), fp32 accumulation.
// Attention double-buffers K/V tiles in LDS via GLOBAL_LOAD_ASYNC_TO_LDS_B128
// (ASYNCcnt) so the DMA overlaps WMMA compute.
// ---------------------------------------------------------------------------

typedef __attribute__((ext_vector_type(16))) __bf16 v16bf;
typedef __attribute__((ext_vector_type(8)))  float  v8f;
typedef __attribute__((ext_vector_type(4)))  int    v4i;

#if defined(__has_builtin)
#if __has_builtin(__builtin_amdgcn_global_load_async_to_lds_b128)
#define HAS_ASYNC_LDS 1
#endif
#endif
#ifndef HAS_ASYNC_LDS
#define HAS_ASYNC_LDS 0
#endif

#if HAS_ASYNC_LDS
typedef __attribute__((address_space(1))) v4i as1_v4i;
typedef __attribute__((address_space(3))) v4i as3_v4i;
__device__ __forceinline__ void async_copy16(const unsigned short* g, unsigned short* l) {
    __builtin_amdgcn_global_load_async_to_lds_b128(
        (as1_v4i*)(void*)g, (as3_v4i*)(void*)l, /*offset=*/0, /*cpol=*/0);
}
__device__ __forceinline__ void async_wait0() {
#if __has_builtin(__builtin_amdgcn_s_wait_asynccnt)
    __builtin_amdgcn_s_wait_asynccnt(0);
#else
    asm volatile("s_wait_asynccnt 0x0" ::: "memory");
#endif
}
#else
__device__ __forceinline__ void async_wait0() {}
#endif

union Frag {
    v16bf        bf;
    unsigned int u[8];
};

__device__ __forceinline__ unsigned short f2bf(float f) {
    unsigned int u = __float_as_uint(f);
    unsigned int r = u + 0x7FFFu + ((u >> 16) & 1u);   // round-to-nearest-even
    return (unsigned short)(r >> 16);
}
__device__ __forceinline__ unsigned int pack2bf(float lo, float hi) {
    return (unsigned int)f2bf(lo) | ((unsigned int)f2bf(hi) << 16);
}

__device__ __forceinline__ v8f wmma_bf16(const Frag& a, const Frag& b, v8f c) {
    return __builtin_amdgcn_wmma_f32_16x16x32_bf16(
        false, a.bf, false, b.bf, (short)0, c, false, false);
}

// ---- fp32 -> bf16 elementwise conversion ----------------------------------
__global__ void f32_to_bf16_kernel(const float* __restrict__ src,
                                   unsigned short* __restrict__ dst, int n) {
    int i = blockIdx.x * blockDim.x + threadIdx.x;
    int stride = gridDim.x * blockDim.x;
    for (; i < n; i += stride) dst[i] = f2bf(src[i]);
}

// ---- fused QKV projection: y = x @ W^T, bf16 out ---------------------------
// wave tile 32x64; jobs: mat(3) x rowTile(256) x colTile(12) = 9216 -> 1152 blocks
// Q,K stored [B,H,S,DH]; V stored transposed [B,H,DH,S].
__global__ void qkv_gemm_kernel(const unsigned short* __restrict__ xb,
                                const unsigned short* __restrict__ wq,
                                const unsigned short* __restrict__ wk,
                                const unsigned short* __restrict__ wv,
                                unsigned short* __restrict__ qws,
                                unsigned short* __restrict__ kws,
                                unsigned short* __restrict__ vtws) {
    const int lane = threadIdx.x & 31;
    const int wave = threadIdx.x >> 5;
    const int job  = blockIdx.x * 8 + wave;
    const int mat  = job / 3072;
    const int rem  = job % 3072;
    const int rowTile = rem / 12;
    const int colTile = rem % 12;
    const unsigned short* W = (mat == 0) ? wq : (mat == 1) ? wk : wv;

    const int  ln = lane & 15;
    const bool lo = lane < 16;
    int koffA[8], koffB[8];
#pragma unroll
    for (int i = 0; i < 8; ++i) {
        koffA[i] = 2 * i + (lo ? 0 : 16);
        koffB[i] = 2 * i + (lo ? 0 : 16);
    }
    const int r0 = rowTile * 32;
    const int c0 = colTile * 64;

    v8f vzero = {};
    v8f acc[8];                      // [half*4 + c]
#pragma unroll
    for (int c = 0; c < 8; ++c) acc[c] = vzero;

    for (int d0 = 0; d0 < 768; d0 += 32) {
        Frag a0, a1;
        const unsigned short* ar0 = xb + (r0 + ln) * 768 + d0;
        const unsigned short* ar1 = xb + (r0 + 16 + ln) * 768 + d0;
#pragma unroll
        for (int i = 0; i < 8; ++i) {
            a0.u[i] = *(const unsigned int*)(ar0 + koffA[i]);
            a1.u[i] = *(const unsigned int*)(ar1 + koffA[i]);
        }
#pragma unroll
        for (int c = 0; c < 4; ++c) {
            Frag b;
            const unsigned short* brow = W + (c0 + c * 16 + ln) * 768 + d0;
#pragma unroll
            for (int i = 0; i < 8; ++i)
                b.u[i] = *(const unsigned int*)(brow + koffB[i]);
            acc[c]     = wmma_bf16(a0, b, acc[c]);
            acc[4 + c] = wmma_bf16(a1, b, acc[4 + c]);
        }
    }
#pragma unroll
    for (int hh = 0; hh < 2; ++hh) {
#pragma unroll
        for (int c = 0; c < 4; ++c) {
#pragma unroll
            for (int i = 0; i < 8; ++i) {
                const int m  = lo ? i : i + 8;
                const int r  = r0 + hh * 16 + m;   // [0, 8192)
                const int b  = r >> 11;
                const int s  = r & 2047;
                const int j  = c0 + c * 16 + ln;
                const int h  = j / 192;
                const int dh = j % 192;
                const unsigned short v = f2bf(acc[hh * 4 + c][i]);
                if (mat == 0)      qws[(((b * 4 + h) * 2048) + s) * 192 + dh] = v;
                else if (mat == 1) kws[(((b * 4 + h) * 2048) + s) * 192 + dh] = v;
                else               vtws[(((b * 4 + h) * 192) + dh) * 2048 + s] = v;
            }
        }
    }
}

// ---- flash attention -------------------------------------------------------
// Block (8 waves) owns one (b,h) and 8 consecutive 16-row q tiles; K/V tiles
// (64 keys) double-buffered in LDS via async-to-LDS DMA overlapped with WMMA.
// logits^T = K·Q^T per 32-key subchunk, online softmax, preout^T += V^T·P^T.
// 256 blocks x 256 threads.
#define KT_STRIDE 200   // 64 rows  (keys)  x 192 d,  padded
#define VT_STRIDE 72    // 192 rows (dh)    x 64 keys, padded
#define KT_ELEMS (64 * KT_STRIDE)
#define VT_ELEMS (192 * VT_STRIDE)

__device__ __forceinline__ void stage_tile(const unsigned short* __restrict__ kbh,
                                           const unsigned short* __restrict__ vbh,
                                           unsigned short* KtBuf, unsigned short* VtBuf,
                                           int j0, int tid) {
    // K rows [j0, j0+64) : 64 x 192, 24 x b128 per row
#pragma unroll
    for (int n = 0; n < 6; ++n) {
        const int idx = tid + n * 256;   // [0, 1536)
        const int row = idx / 24, seg = idx % 24;
        const unsigned short* src = kbh + (j0 + row) * 192 + seg * 8;
        unsigned short*       dst = KtBuf + row * KT_STRIDE + seg * 8;
#if HAS_ASYNC_LDS
        async_copy16(src, dst);
#else
        *(uint4*)dst = *(const uint4*)src;
#endif
    }
    // V^T cols [j0, j0+64) : 192 x 64, 8 x b128 per row
#pragma unroll
    for (int n = 0; n < 6; ++n) {
        const int idx = tid + n * 256;   // [0, 1536)
        const int row = idx / 8, seg = idx % 8;
        const unsigned short* src = vbh + row * 2048 + j0 + seg * 8;
        unsigned short*       dst = VtBuf + row * VT_STRIDE + seg * 8;
#if HAS_ASYNC_LDS
        async_copy16(src, dst);
#else
        *(uint4*)dst = *(const uint4*)src;
#endif
    }
}

__global__ void attn_kernel(const unsigned short* __restrict__ qws,
                            const unsigned short* __restrict__ kws,
                            const unsigned short* __restrict__ vtws,
                            unsigned short* __restrict__ preb) {
    __shared__ unsigned short Kt[2][KT_ELEMS];   // 2 x 25600 B
    __shared__ unsigned short Vt[2][VT_ELEMS];   // 2 x 27648 B

    const int lane = threadIdx.x & 31;
    const int wave = threadIdx.x >> 5;
    const int tid  = threadIdx.x;

    const int bh = blockIdx.x >> 4;          // [0,16)
    const int tb = blockIdx.x & 15;          // 8-tile group within (b,h)
    const int b  = bh >> 2;
    const int h  = bh & 3;
    const int i0w   = (tb * 8 + wave) * 16;  // this wave's q-tile start
    const int i0max = (tb * 8 + 7) * 16;     // block-uniform loop bound

    const unsigned short* qbh = qws  + (size_t)bh * 2048 * 192;
    const unsigned short* kbh = kws  + (size_t)bh * 2048 * 192;
    const unsigned short* vbh = vtws + (size_t)bh * 192 * 2048;

    const int  ln = lane & 15;
    const bool lo = lane < 16;
    int koffA[8], koffB[8];
#pragma unroll
    for (int i = 0; i < 8; ++i) {
        koffA[i] = 2 * i + (lo ? 0 : 16);
        koffB[i] = 2 * i + (lo ? 0 : 16);
    }

    // Q fragments (B layout: n = q row, K = d chunk), held for whole tile.
    Frag qf[6];
#pragma unroll
    for (int dc = 0; dc < 6; ++dc) {
        const unsigned short* qrow = qbh + (i0w + ln) * 192 + dc * 32;
#pragma unroll
        for (int i = 0; i < 8; ++i)
            qf[dc].u[i] = *(const unsigned int*)(qrow + koffB[i]);
    }

    v8f vzero = {};
    v8f acc[12];
#pragma unroll
    for (int t = 0; t < 12; ++t) acc[t] = vzero;

    float mrow = -1e30f, lsum = 0.f;
    const float sc = 0.07216878364870323f;   // 1/sqrt(192)

    // prologue: stage tile 0 into buffer 0
    stage_tile(kbh, vbh, Kt[0], Vt[0], 0, tid);

    for (int j0 = 0; j0 <= i0max; j0 += 64) {
        const int cur = (j0 >> 6) & 1;
        async_wait0();                       // my tile-j0 loads complete
        __syncthreads();                     // everyone's loads done; prev compute done
        if (j0 + 64 <= i0max)                // block-uniform: prefetch next tile
            stage_tile(kbh, vbh, Kt[cur ^ 1], Vt[cur ^ 1], j0 + 64, tid);

        if (j0 > i0w) continue;              // wave-uniform: no compute this tile
        const unsigned short* KtC = Kt[cur];
        const unsigned short* VtC = Vt[cur];

#pragma unroll
        for (int sub = 0; sub < 2; ++sub) {
            const int jj = j0 + sub * 32;    // 32-key subchunk
            if (jj > i0w) continue;          // wave-uniform
            const int lk = sub * 32;         // local key base in LDS

            v8f lt0 = vzero, lt1 = vzero;
#pragma unroll
            for (int dc = 0; dc < 6; ++dc) {
                Frag kf;
                const unsigned short* krow = KtC + (lk + ln) * KT_STRIDE + dc * 32;
#pragma unroll
                for (int i = 0; i < 8; ++i)
                    kf.u[i] = *(const unsigned int*)(krow + koffA[i]);
                lt0 = wmma_bf16(kf, qf[dc], lt0);
            }
#pragma unroll
            for (int dc = 0; dc < 6; ++dc) {
                Frag kf;
                const unsigned short* krow = KtC + (lk + 16 + ln) * KT_STRIDE + dc * 32;
#pragma unroll
                for (int i = 0; i < 8; ++i)
                    kf.u[i] = *(const unsigned int*)(krow + koffA[i]);
                lt1 = wmma_bf16(kf, qf[dc], lt1);
            }

            // scale + causal mask (lt tiles: M=key in vgprs, N=qrow in lanes)
            const bool needMask = (jj + 31) > i0w;
            float vals[16];
#pragma unroll
            for (int i = 0; i < 8; ++i) { vals[i] = lt0[i] * sc; vals[8 + i] = lt1[i] * sc; }
            if (needMask) {
                const int qr = i0w + ln;
#pragma unroll
                for (int i = 0; i < 8; ++i) {
                    const int k0 = jj + i + (lo ? 0 : 8);
                    const int k1 = jj + 16 + i + (lo ? 0 : 8);
                    if (k0 > qr) vals[i]     = -1e30f;
                    if (k1 > qr) vals[8 + i] = -1e30f;
                }
            }
            // online softmax over this subchunk's keys
            float cmax = vals[0];
#pragma unroll
            for (int i = 1; i < 16; ++i) cmax = fmaxf(cmax, vals[i]);
            cmax = fmaxf(cmax, __shfl_xor(cmax, 16, 32));
            const float mnew = fmaxf(mrow, cmax);
            float psum = 0.f;
#pragma unroll
            for (int i = 0; i < 16; ++i) { vals[i] = __expf(vals[i] - mnew); psum += vals[i]; }
            psum += __shfl_xor(psum, 16, 32);
            const float resc = __expf(mrow - mnew);
            lsum = lsum * resc + psum;
            mrow = mnew;
#pragma unroll
            for (int t = 0; t < 12; ++t) acc[t] = acc[t] * resc;

            // build P^T B-fragment (K = 32 keys, N = 16 q rows)
            float othr[16];
#pragma unroll
            for (int i = 0; i < 16; ++i) othr[i] = __shfl_xor(vals[i], 16, 32);
            Frag pf;
#pragma unroll
            for (int i = 0; i < 8; ++i) {
                float a0, a1;
                if (i < 4) {
                    a0 = lo ? vals[2 * i]     : othr[8 + 2 * i];
                    a1 = lo ? vals[2 * i + 1] : othr[8 + 2 * i + 1];
                } else {
                    const int i4 = i - 4;
                    a0 = lo ? othr[2 * i4]     : vals[8 + 2 * i4];
                    a1 = lo ? othr[2 * i4 + 1] : vals[8 + 2 * i4 + 1];
                }
                pf.u[i] = pack2bf(a0, a1);
            }

            // preout^T += V^T(16dh x 32keys) · P^T(32keys x 16q)
#pragma unroll
            for (int t = 0; t < 12; ++t) {
                Frag vf;
                const unsigned short* vrow = VtC + (t * 16 + ln) * VT_STRIDE + lk;
#pragma unroll
                for (int i = 0; i < 8; ++i)
                    vf.u[i] = *(const unsigned int*)(vrow + koffA[i]);
                acc[t] = wmma_bf16(vf, pf, acc[t]);
            }
        }
    }

    const float inv = 1.0f / lsum;
    // preout^T D-tiles: M=dh in vgprs, N=q in lanes -> store [B,S,H*DH] bf16
#pragma unroll
    for (int t = 0; t < 12; ++t) {
#pragma unroll
        for (int i = 0; i < 8; ++i) {
            const int dh = t * 16 + (lo ? i : i + 8);
            const int q  = i0w + ln;
            preb[((size_t)(b * 2048 + q)) * 768 + h * 192 + dh] = f2bf(acc[t][i] * inv);
        }
    }
}

// ---- output projection: out = preout @ Wo^T, fp32 out ----------------------
// wave tile 32x64; 3072 jobs -> 384 blocks
__global__ void out_gemm_kernel(const unsigned short* __restrict__ preb,
                                const unsigned short* __restrict__ wo,
                                float* __restrict__ out) {
    const int lane = threadIdx.x & 31;
    const int wave = threadIdx.x >> 5;
    const int job  = blockIdx.x * 8 + wave;
    const int rowTile = job / 12;
    const int colTile = job % 12;

    const int  ln = lane & 15;
    const bool lo = lane < 16;
    int koffA[8], koffB[8];
#pragma unroll
    for (int i = 0; i < 8; ++i) {
        koffA[i] = 2 * i + (lo ? 0 : 16);
        koffB[i] = 2 * i + (lo ? 0 : 16);
    }
    const int r0 = rowTile * 32;
    const int c0 = colTile * 64;

    v8f vzero = {};
    v8f acc[8];
#pragma unroll
    for (int c = 0; c < 8; ++c) acc[c] = vzero;

    for (int d0 = 0; d0 < 768; d0 += 32) {
        Frag a0, a1;
        const unsigned short* ar0 = preb + (r0 + ln) * 768 + d0;
        const unsigned short* ar1 = preb + (r0 + 16 + ln) * 768 + d0;
#pragma unroll
        for (int i = 0; i < 8; ++i) {
            a0.u[i] = *(const unsigned int*)(ar0 + koffA[i]);
            a1.u[i] = *(const unsigned int*)(ar1 + koffA[i]);
        }
#pragma unroll
        for (int c = 0; c < 4; ++c) {
            Frag b;
            const unsigned short* brow = wo + (c0 + c * 16 + ln) * 768 + d0;
#pragma unroll
            for (int i = 0; i < 8; ++i)
                b.u[i] = *(const unsigned int*)(brow + koffB[i]);
            acc[c]     = wmma_bf16(a0, b, acc[c]);
            acc[4 + c] = wmma_bf16(a1, b, acc[4 + c]);
        }
    }
#pragma unroll
    for (int hh = 0; hh < 2; ++hh) {
#pragma unroll
        for (int c = 0; c < 4; ++c) {
#pragma unroll
            for (int i = 0; i < 8; ++i) {
                const int m = lo ? i : i + 8;
                const int r = r0 + hh * 16 + m;
                const int j = c0 + c * 16 + ln;
                out[(size_t)r * 768 + j] = acc[hh * 4 + c][i];
            }
        }
    }
}

// ---------------------------------------------------------------------------
extern "C" void kernel_launch(void* const* d_in, const int* in_sizes, int n_in,
                              void* d_out, int out_size, void* d_ws, size_t ws_size,
                              hipStream_t stream) {
    (void)in_sizes; (void)n_in; (void)out_size; (void)ws_size;
    const float* x  = (const float*)d_in[0];
    const float* Wq = (const float*)d_in[1];
    const float* Wk = (const float*)d_in[2];
    const float* Wv = (const float*)d_in[3];
    const float* Wo = (const float*)d_in[4];
    float* out = (float*)d_out;

    char* ws = (char*)d_ws;
    size_t off = 0;
    auto alloc = [&](size_t bytes) -> void* {
        void* p = ws + off;
        off += (bytes + 255) & ~(size_t)255;
        return p;
    };
    const size_t XN = (size_t)8192 * 768;   // B*S*D
    const size_t WN = (size_t)768 * 768;

    unsigned short* xb   = (unsigned short*)alloc(XN * 2);
    unsigned short* wqb  = (unsigned short*)alloc(WN * 2);
    unsigned short* wkb  = (unsigned short*)alloc(WN * 2);
    unsigned short* wvb  = (unsigned short*)alloc(WN * 2);
    unsigned short* wob  = (unsigned short*)alloc(WN * 2);
    unsigned short* qws  = (unsigned short*)alloc(XN * 2);  // [B,H,S,DH]
    unsigned short* kws  = (unsigned short*)alloc(XN * 2);  // [B,H,S,DH]
    unsigned short* vtws = (unsigned short*)alloc(XN * 2);  // [B,H,DH,S]
    unsigned short* preb = (unsigned short*)alloc(XN * 2);  // [B,S,D]

    f32_to_bf16_kernel<<<512, 256, 0, stream>>>(x,  xb,  (int)XN);
    f32_to_bf16_kernel<<<256, 256, 0, stream>>>(Wq, wqb, (int)WN);
    f32_to_bf16_kernel<<<256, 256, 0, stream>>>(Wk, wkb, (int)WN);
    f32_to_bf16_kernel<<<256, 256, 0, stream>>>(Wv, wvb, (int)WN);
    f32_to_bf16_kernel<<<256, 256, 0, stream>>>(Wo, wob, (int)WN);

    qkv_gemm_kernel<<<1152, 256, 0, stream>>>(xb, wqb, wkb, wvb, qws, kws, vtws);
    attn_kernel<<<256, 256, 0, stream>>>(qws, kws, vtws, preb);
    out_gemm_kernel<<<384, 256, 0, stream>>>(preb, wob, out);
}
